// MDTA_953482739943
// MI455X (gfx1250) — matrix operand order repaired and checked by reading, not compiled, not run on hardware
//
#include <hip/hip_runtime.h>

#define HEADS 4
#define Bn 8
#define Cc 192
#define C3 576
#define Himg 128
#define Wimg 128
#define HW 16384
#define Dh 48
#define KSPLIT 16

typedef __attribute__((ext_vector_type(16))) __bf16 v16bf;
typedef __attribute__((ext_vector_type(8)))  float  v8f;

union FragAB { v16bf v; __bf16 e[16]; };
union FragC  { v8f   v; float  e[8]; };
union Pack8  { uint4 u; __bf16 h[8]; };

__device__ __forceinline__ __bf16 f2bf(float f)  { return (__bf16)f; }
__device__ __forceinline__ float  bf2f(__bf16 h) { return (float)h; }

__device__ __forceinline__ v8f wmma_bf16(v16bf a, v16bf b, v8f c) {
  // D = A(16x32 bf16) * B(32x16 bf16) + C(16x16 f32)
  return __builtin_amdgcn_wmma_f32_16x16x32_bf16(false, a, false, b, (short)0, c,
                                                 false, false);
}

// ---------------------------------------------------------------------------
// K0: convert 1x1-conv weights to bf16
// ---------------------------------------------------------------------------
__global__ void k_prep(const float* __restrict__ qkv_w,
                       const float* __restrict__ proj_w,
                       __bf16* __restrict__ wq, __bf16* __restrict__ wp) {
  int i = blockIdx.x * 256 + threadIdx.x;
  if (i < C3 * Cc) wq[i] = f2bf(qkv_w[i]);
  if (i < Cc * Cc) wp[i] = f2bf(proj_w[i]);
}

// ---------------------------------------------------------------------------
// K1: qkv = Wqkv(576x192) * x(192xHW) + bias -> bf16.  grid(HW/64, 1, B)
//     Full-K B panel staged ONCE per block (x read from HBM exactly once);
//     loop over all 9 M-tiles with per-tile A panel. No barriers in K loop.
// ---------------------------------------------------------------------------
__global__ __launch_bounds__(128)
void k_gemm_qkv(const float* __restrict__ x, const __bf16* __restrict__ wq,
                const float* __restrict__ bias, __bf16* __restrict__ out) {
  __shared__ __attribute__((aligned(64))) __bf16 lB[64][208];  // [n][k] padded
  __shared__ __attribute__((aligned(64))) __bf16 lA[64][192];  // [m][k]
  const int tid = threadIdx.x, lane = tid & 31, wave = tid >> 5;
  const int wy = wave >> 1, wx = wave & 1;
  const int n0 = blockIdx.x * 64, b = blockIdx.z;
  const float* xb = x + (size_t)b * Cc * HW;
  const int mrow = lane & 15, ks8 = (lane >> 4) * 8, ks16 = (lane >> 4) * 16;
  const int nl = lane & 15, mh = (lane >> 4) * 8;

  // ---- stage full-K B panel (f32 -> bf16, transposed): 192k x 64n ----
  for (int i = tid; i < 3072; i += 128) {        // chunks of 4 floats (b128)
    int r = i >> 4;                              // k row 0..191
    int c0 = (i & 15) * 4;                       // n col
    float4 v = *(const float4*)(xb + (size_t)r * HW + n0 + c0);
    lB[c0 + 0][r] = f2bf(v.x);
    lB[c0 + 1][r] = f2bf(v.y);
    lB[c0 + 2][r] = f2bf(v.z);
    lB[c0 + 3][r] = f2bf(v.w);
  }

  for (int m0 = 0; m0 < C3; m0 += 64) {
    __syncthreads();  // lA WAR vs previous tile; first iter: publish lB
    // ---- stage A panel: 64m x 192k bf16, b128 loads ----
    for (int i = tid; i < 1536; i += 128) {      // chunks of 8 bf16
      int r = i / 24;
      int c0 = (i % 24) * 8;
      *(uint4*)&lA[r][c0] =
          *(const uint4*)(wq + (size_t)(m0 + r) * Cc + c0);
    }
    __syncthreads();

    FragC acc[2][2] = {};
    for (int k0 = 0; k0 < Cc; k0 += 32) {
      FragAB af[2], bfr[2];
      for (int fm = 0; fm < 2; ++fm) {
        int m = wy * 32 + fm * 16 + mrow;
#pragma unroll
        for (int e = 0; e < 8; ++e) {
          af[fm].e[e]     = lA[m][k0 + ks16 + e];
          af[fm].e[e + 8] = lA[m][k0 + ks16 + 8 + e];
        }
      }
      for (int fn = 0; fn < 2; ++fn) {
        int n = wx * 32 + fn * 16 + mrow;
        bfr[fn].v = *(const v16bf*)&lB[n][k0 + ks16];
      }
#pragma unroll
      for (int fm = 0; fm < 2; ++fm)
#pragma unroll
        for (int fn = 0; fn < 2; ++fn)
          acc[fm][fn].v = wmma_bf16(af[fm].v, bfr[fn].v, acc[fm][fn].v);
    }

    for (int fm = 0; fm < 2; ++fm)
      for (int fn = 0; fn < 2; ++fn) {
        int gn = n0 + wx * 32 + fn * 16 + nl;
#pragma unroll
        for (int r = 0; r < 8; ++r) {
          int gm = m0 + wy * 32 + fm * 16 + mh + r;
          out[((size_t)b * C3 + gm) * HW + gn] =
              f2bf(acc[fm][fn].e[r] + bias[gm]);
        }
      }
  }
}

// ---------------------------------------------------------------------------
// K2: depthwise 3x3 (pad 1) + bias; also per-tile sum-of-squares partials for
//     q/k channels (deterministic reduction later). grid(8,8,B*576) block(16,16)
// ---------------------------------------------------------------------------
__global__ __launch_bounds__(256)
void k_dwconv(const __bf16* __restrict__ qpre, const float* __restrict__ dww,
              const float* __restrict__ dwb, __bf16* __restrict__ qd,
              float* __restrict__ nsp) {
  __shared__ float tile[18][18];
  __shared__ float red[256];
  const int bc = blockIdx.z, b = bc / C3, ch = bc % C3;
  const int x0 = blockIdx.x * 16, y0 = blockIdx.y * 16;
  const int tx = threadIdx.x, ty = threadIdx.y, tid = ty * 16 + tx;
  const __bf16* src = qpre + (size_t)bc * HW;
  for (int i = tid; i < 18 * 18; i += 256) {
    int col = i % 18, rowi = i / 18;
    int gx = x0 + col - 1, gy = y0 + rowi - 1;
    float v = 0.f;
    if (gx >= 0 && gx < Wimg && gy >= 0 && gy < Himg)
      v = bf2f(src[gy * Wimg + gx]);
    tile[rowi][col] = v;
  }
  __syncthreads();
  const float* w = dww + ch * 9;
  float s = dwb[ch];
#pragma unroll
  for (int ky = 0; ky < 3; ++ky)
#pragma unroll
    for (int kx = 0; kx < 3; ++kx)
      s += w[ky * 3 + kx] * tile[ty + ky][tx + kx];
  __bf16 hv = f2bf(s);
  qd[(size_t)bc * HW + (y0 + ty) * Wimg + (x0 + tx)] = hv;
  if (ch < 2 * Cc) {  // q or k channel: accumulate ||row||^2 partial
    float v = bf2f(hv);
    red[tid] = v * v;
    __syncthreads();
    for (int st = 128; st > 0; st >>= 1) {
      if (tid < st) red[tid] += red[tid + st];
      __syncthreads();
    }
    if (tid == 0)
      nsp[(size_t)(b * 2 * Cc + ch) * 64 + blockIdx.y * 8 + blockIdx.x] = red[0];
  }
}

// ---------------------------------------------------------------------------
// K3: raw Gram S = q * k^T (K = HW, split KSPLIT ways). grid(KSPLIT, heads, B)
//     Each of 4 waves accumulates full 48x48 over its n-range; fixed-order
//     reduction into a per-ksplit slice (deterministic, no atomics).
// ---------------------------------------------------------------------------
__global__ __launch_bounds__(128)
void k_gram(const __bf16* __restrict__ qd, float* __restrict__ Sg) {
  __shared__ __attribute__((aligned(64))) float sSw[4][Dh * Dh];  // 36 KB
  __bf16* lT = (__bf16*)sSw;  // aliased: per-wave 96x32 bf16 tiles (24 KB)
  const int tid = threadIdx.x, lane = tid & 31, wave = tid >> 5;
  const int ks = blockIdx.x, h = blockIdx.y, b = blockIdx.z;
  const int chunk = HW / KSPLIT / 4;                 // 256 n per wave
  const int nbase = ks * (HW / KSPLIT) + wave * chunk;
  const __bf16* qbase = qd + ((size_t)b * C3 + h * Dh) * HW;
  const __bf16* kbp   = qd + ((size_t)b * C3 + Cc + h * Dh) * HW;
  const int mrow = lane & 15, koff = (lane >> 4) * 8, kb = (lane >> 4) * 16;
  __bf16* myT = lT + wave * 96 * 32;
  FragC acc[3][3] = {};
  for (int n0 = nbase; n0 < nbase + chunk; n0 += 32) {
    for (int rr = lane; rr < 96; rr += 32) {  // 96 rows x 32 bf16 (64B each)
      const __bf16* rp = (rr < Dh) ? (qbase + (size_t)rr * HW + n0)
                                   : (kbp + (size_t)(rr - Dh) * HW + n0);
      const uint4* s4 = (const uint4*)rp;
      uint4* d4 = (uint4*)(myT + rr * 32);
      d4[0] = s4[0]; d4[1] = s4[1]; d4[2] = s4[2]; d4[3] = s4[3];
    }
    __syncthreads();
    FragAB af[3], bfr[3];
    for (int f = 0; f < 3; ++f) {
      const __bf16* qr = myT + (f * 16 + mrow) * 32;
#pragma unroll
      for (int e = 0; e < 8; ++e) {
        af[f].e[e]     = qr[kb + e];
        af[f].e[e + 8] = qr[kb + 8 + e];
      }
      bfr[f].v = *(const v16bf*)(myT + (Dh + f * 16 + mrow) * 32 + kb);
    }
#pragma unroll
    for (int fi = 0; fi < 3; ++fi)
#pragma unroll
      for (int fj = 0; fj < 3; ++fj)
        acc[fi][fj].v = wmma_bf16(af[fi].v, bfr[fj].v, acc[fi][fj].v);
  }
  __syncthreads();
  const int nl = lane & 15, mh = (lane >> 4) * 8;
  for (int fi = 0; fi < 3; ++fi)
    for (int fj = 0; fj < 3; ++fj)
#pragma unroll
      for (int r = 0; r < 8; ++r)
        sSw[wave][(fi * 16 + mh + r) * Dh + fj * 16 + nl] = acc[fi][fj].e[r];
  __syncthreads();
  float* dst = Sg + ((size_t)(ks * Bn * HEADS) + b * HEADS + h) * (Dh * Dh);
  for (int i = tid; i < Dh * Dh; i += 128)
    dst[i] = sSw[0][i] + sSw[1][i] + sSw[2][i] + sSw[3][i];
}

// ---------------------------------------------------------------------------
// K3b: fold norms + temperature, softmax over j, store attn bf16 padded 48x64.
//      grid(B*heads) block(64)
// ---------------------------------------------------------------------------
__global__ void k_softmax(const float* __restrict__ Sg,
                          const float* __restrict__ nsp,
                          const float* __restrict__ temp,
                          __bf16* __restrict__ attn) {
  __shared__ float invK[Dh];
  const int bh = blockIdx.x, b = bh >> 2, h = bh & 3;
  const int i = threadIdx.x;
  if (i < Dh) {
    float s = 0.f;
    const float* p = nsp + (size_t)(b * 2 * Cc + Cc + h * Dh + i) * 64;
    for (int k = 0; k < 64; ++k) s += p[k];
    invK[i] = 1.f / fmaxf(sqrtf(s), 1e-12f);
  }
  __syncthreads();
  if (i >= Dh) return;
  float sq = 0.f;
  const float* pq = nsp + (size_t)(b * 2 * Cc + h * Dh + i) * 64;
  for (int k = 0; k < 64; ++k) sq += pq[k];
  const float invq = 1.f / fmaxf(sqrtf(sq), 1e-12f);
  const float t = temp[h];
  float row[Dh];
  float m = -3.0e38f;
  for (int j = 0; j < Dh; ++j) {
    float s = 0.f;
    for (int ks = 0; ks < KSPLIT; ++ks)
      s += Sg[((size_t)(ks * Bn * HEADS) + bh) * (Dh * Dh) + i * Dh + j];
    float v = s * invq * invK[j] * t;
    row[j] = v;
    m = fmaxf(m, v);
  }
  float sum = 0.f;
  for (int j = 0; j < Dh; ++j) { row[j] = expf(row[j] - m); sum += row[j]; }
  const float inv = 1.f / sum;
  __bf16* dst = attn + (size_t)bh * Dh * 64 + (size_t)i * 64;
  for (int j = 0; j < Dh; ++j)  dst[j] = f2bf(row[j] * inv);
  for (int j = Dh; j < 64; ++j) dst[j] = f2bf(0.f);
}

// ---------------------------------------------------------------------------
// K4: out = attn(48x48, K padded to 64) * v(48xHW) -> bf16. grid(HW/64,heads,B)
// ---------------------------------------------------------------------------
__global__ __launch_bounds__(128)
void k_av(const __bf16* __restrict__ attn, const __bf16* __restrict__ qd,
          __bf16* __restrict__ aout) {
  __shared__ __attribute__((aligned(64))) __bf16 lA[Dh][64];  // attn
  __shared__ __attribute__((aligned(64))) __bf16 lV[64][80];  // [n][k] padded
  const int tid = threadIdx.x, lane = tid & 31, wave = tid >> 5;
  const int n0 = blockIdx.x * 64, h = blockIdx.y, b = blockIdx.z;
  const __bf16* A = attn + (size_t)(b * HEADS + h) * Dh * 64;
  const __bf16* V = qd + ((size_t)b * C3 + 2 * Cc + h * Dh) * HW;
  for (int i = tid; i < 384; i += 128)  // 3072 bf16 as b128 chunks
    *(uint4*)((__bf16*)lA + i * 8) = *(const uint4*)(A + i * 8);
  for (int i = tid; i < 512; i += 128) {  // 64 rows x 8 chunks of 8 bf16
    int r = i >> 3, c0 = (i & 7) * 8;
    Pack8 p;
    if (r < Dh) p.u = *(const uint4*)(V + (size_t)r * HW + n0 + c0);
    else { p.u.x = 0; p.u.y = 0; p.u.z = 0; p.u.w = 0; }
#pragma unroll
    for (int j = 0; j < 8; ++j) lV[c0 + j][r] = p.h[j];
  }
  __syncthreads();
  const int mrow = lane & 15;
  FragC acc[3] = {};
#pragma unroll
  for (int kk0 = 0; kk0 < 64; kk0 += 32) {
    const int koff = kk0 + (lane >> 4) * 8;
    const int kb   = kk0 + (lane >> 4) * 16;
    FragAB bfr;
    bfr.v = *(const v16bf*)&lV[wave * 16 + mrow][kb];
#pragma unroll
    for (int f = 0; f < 3; ++f) {
      FragAB af;
#pragma unroll
      for (int e = 0; e < 8; ++e) {
        af.e[e]     = lA[f * 16 + mrow][kb + e];
        af.e[e + 8] = lA[f * 16 + mrow][kb + 8 + e];
      }
      acc[f].v = wmma_bf16(af.v, bfr.v, acc[f].v);
    }
  }
  const int nl = lane & 15, mh = (lane >> 4) * 8;
  __bf16* dst = aout + (size_t)b * Cc * HW;
  for (int f = 0; f < 3; ++f)
#pragma unroll
    for (int r = 0; r < 8; ++r) {
      int chm = h * Dh + f * 16 + mh + r;
      dst[(size_t)chm * HW + n0 + wave * 16 + nl] = f2bf(acc[f].e[r]);
    }
}

// ---------------------------------------------------------------------------
// K5: final = Wp(192x192) * aout(192xHW) + bias -> f32. grid(HW/64, 1, B)
//     Same full-K panel structure as K1 (B staged once, 3 M-tiles).
// ---------------------------------------------------------------------------
__global__ __launch_bounds__(128)
void k_gemm_proj(const __bf16* __restrict__ ain, const __bf16* __restrict__ wp,
                 const float* __restrict__ bias, float* __restrict__ out) {
  __shared__ __attribute__((aligned(64))) __bf16 lB[64][208];  // [n][k] padded
  __shared__ __attribute__((aligned(64))) __bf16 lA[64][192];  // [m][k]
  const int tid = threadIdx.x, lane = tid & 31, wave = tid >> 5;
  const int wy = wave >> 1, wx = wave & 1;
  const int n0 = blockIdx.x * 64, b = blockIdx.z;
  const __bf16* ab = ain + (size_t)b * Cc * HW;
  const int mrow = lane & 15, ks8 = (lane >> 4) * 8, ks16 = (lane >> 4) * 16;
  const int nl = lane & 15, mh = (lane >> 4) * 8;

  // ---- stage full-K B panel (bf16, transposed): 192k x 64n ----
  for (int i = tid; i < 1536; i += 128) {        // chunks of 8 bf16 (b128)
    int r = i >> 3;                              // k row 0..191
    int c0 = (i & 7) * 8;                        // n col
    Pack8 p;
    p.u = *(const uint4*)(ab + (size_t)r * HW + n0 + c0);
#pragma unroll
    for (int j = 0; j < 8; ++j) lB[c0 + j][r] = p.h[j];
  }

  for (int m0 = 0; m0 < Cc; m0 += 64) {
    __syncthreads();
    for (int i = tid; i < 1536; i += 128) {      // A panel, b128 loads
      int r = i / 24;
      int c0 = (i % 24) * 8;
      *(uint4*)&lA[r][c0] =
          *(const uint4*)(wp + (size_t)(m0 + r) * Cc + c0);
    }
    __syncthreads();

    FragC acc[2][2] = {};
    for (int k0 = 0; k0 < Cc; k0 += 32) {
      FragAB af[2], bfr[2];
      for (int fm = 0; fm < 2; ++fm) {
        int m = wy * 32 + fm * 16 + mrow;
#pragma unroll
        for (int e = 0; e < 8; ++e) {
          af[fm].e[e]     = lA[m][k0 + ks16 + e];
          af[fm].e[e + 8] = lA[m][k0 + ks16 + 8 + e];
        }
      }
      for (int fn = 0; fn < 2; ++fn) {
        int n = wx * 32 + fn * 16 + mrow;
        bfr[fn].v = *(const v16bf*)&lB[n][k0 + ks16];
      }
#pragma unroll
      for (int fm = 0; fm < 2; ++fm)
#pragma unroll
        for (int fn = 0; fn < 2; ++fn)
          acc[fm][fn].v = wmma_bf16(af[fm].v, bfr[fn].v, acc[fm][fn].v);
    }

    for (int fm = 0; fm < 2; ++fm)
      for (int fn = 0; fn < 2; ++fn) {
        int gn = n0 + wx * 32 + fn * 16 + nl;
#pragma unroll
        for (int r = 0; r < 8; ++r) {
          int gm = m0 + wy * 32 + fm * 16 + mh + r;
          out[((size_t)b * Cc + gm) * HW + gn] = acc[fm][fn].e[r] + bias[gm];
        }
      }
  }
}

// ---------------------------------------------------------------------------
extern "C" void kernel_launch(void* const* d_in, const int* in_sizes, int n_in,
                              void* d_out, int out_size, void* d_ws,
                              size_t ws_size, hipStream_t stream) {
  (void)in_sizes; (void)n_in; (void)out_size; (void)ws_size;
  const float* x      = (const float*)d_in[0];
  const float* qkv_w  = (const float*)d_in[1];
  const float* qkv_b  = (const float*)d_in[2];
  const float* dw_w   = (const float*)d_in[3];
  const float* dw_b   = (const float*)d_in[4];
  const float* temp   = (const float*)d_in[5];
  const float* proj_w = (const float*)d_in[6];
  const float* proj_b = (const float*)d_in[7];
  float* out = (float*)d_out;

  char* ws = (char*)d_ws;
  size_t off = 0;
  auto take = [&](size_t bytes) {
    size_t r = off;
    off += (bytes + 255) & ~(size_t)255;
    return r;
  };
  __bf16* wq   = (__bf16*)(ws + take((size_t)C3 * Cc * 2));
  __bf16* wp   = (__bf16*)(ws + take((size_t)Cc * Cc * 2));
  float*  nsp  = (float*)(ws + take((size_t)Bn * 2 * Cc * 64 * 4));
  float*  Sg   = (float*)(ws + take((size_t)KSPLIT * Bn * HEADS * Dh * Dh * 4));
  __bf16* attn = (__bf16*)(ws + take((size_t)Bn * HEADS * Dh * 64 * 2));
  __bf16* qpre = (__bf16*)(ws + take((size_t)Bn * C3 * HW * 2));
  __bf16* qd   = (__bf16*)(ws + take((size_t)Bn * C3 * HW * 2));
  __bf16* aout = qpre;  // qpre is dead after k_dwconv; reuse for attn output

  k_prep<<<dim3((C3 * Cc + 255) / 256), 256, 0, stream>>>(qkv_w, proj_w, wq, wp);
  k_gemm_qkv<<<dim3(HW / 64, 1, Bn), 128, 0, stream>>>(x, wq, qkv_b, qpre);
  k_dwconv<<<dim3(Wimg / 16, Himg / 16, Bn * C3), dim3(16, 16), 0, stream>>>(
      qpre, dw_w, dw_b, qd, nsp);
  k_gram<<<dim3(KSPLIT, HEADS, Bn), 128, 0, stream>>>(qd, Sg);
  k_softmax<<<dim3(Bn * HEADS), 64, 0, stream>>>(Sg, nsp, temp, attn);
  k_av<<<dim3(HW / 64, HEADS, Bn), 128, 0, stream>>>(attn, qd, aout);
  k_gemm_proj<<<dim3(HW / 64, 1, Bn), 128, 0, stream>>>(aout, wp, proj_b, out);
}
